// MaxwellGeometricGNNBlock_86088324481808
// MI455X (gfx1250) — compile-verified, matrix operand order checked by reading, not run on hardware
//
#include <hip/hip_runtime.h>
#include <math.h>

// ---------------------------------------------------------------------------
// CDNA5 WMMA / TDM types
// ---------------------------------------------------------------------------
typedef __attribute__((ext_vector_type(16))) __bf16 v16bf;
typedef __attribute__((ext_vector_type(8)))  __bf16 v8bf;
typedef __attribute__((ext_vector_type(8)))  float  v8f;
typedef __attribute__((ext_vector_type(4)))  unsigned int u32x4;
typedef __attribute__((ext_vector_type(8)))  int    i32x8;
typedef __attribute__((ext_vector_type(4)))  int    i32x4;

#define HID   128
#define HEADS 4
#define CHD   32

#if defined(__has_builtin)
#  if __has_builtin(__builtin_amdgcn_tensor_load_to_lds)
#    define HAVE_TDM 1
#  endif
#endif
#ifndef HAVE_TDM
#  define HAVE_TDM 0
#endif

// ---------------------------------------------------------------------------
// Fragment loaders (CDNA5 ISA 7.12.2):
//  A/B (16x32 / 32x16 bf16): lane L holds row/col = L&15; K elements
//  {k0..k0+7, k0+16..k0+23}, k0 = 32*ks + ((L&16)?8:0).
// ---------------------------------------------------------------------------
__device__ __forceinline__ v16bf load_a_frag_f32(const float* __restrict__ Arow,
                                                 const float* __restrict__ abias,
                                                 int ks, int lane) {
  int k0 = ks * 32 + ((lane & 16) ? 8 : 0);
  v16bf a;
  if (abias) {
#pragma unroll
    for (int i = 0; i < 8; ++i) a[i]     = (__bf16)(Arow[k0 + i]      + abias[k0 + i]);
#pragma unroll
    for (int i = 0; i < 8; ++i) a[8 + i] = (__bf16)(Arow[k0 + 16 + i] + abias[k0 + 16 + i]);
  } else {
#pragma unroll
    for (int i = 0; i < 8; ++i) a[i]     = (__bf16)Arow[k0 + i];
#pragma unroll
    for (int i = 0; i < 8; ++i) a[8 + i] = (__bf16)Arow[k0 + 16 + i];
  }
  return a;
}

// bf16 source (global or LDS): two 16-byte vector loads, no conversion.
__device__ __forceinline__ v16bf load_frag_bf16(const __bf16* __restrict__ p,
                                                int ks, int lane) {
  int k0 = ks * 32 + ((lane & 16) ? 8 : 0);
  v8bf lo = *(const v8bf*)(p + k0);
  v8bf hi = *(const v8bf*)(p + k0 + 16);
  return __builtin_shufflevector(lo, hi, 0, 1, 2, 3, 4, 5, 6, 7,
                                 8, 9, 10, 11, 12, 13, 14, 15);
}

// ---------------------------------------------------------------------------
// Weight convert + transpose to bf16 column-major: Wt[n*K + k] = bf16(W[k*Ncols+n])
// ---------------------------------------------------------------------------
__global__ void transpose_bf16_kernel(const float* __restrict__ W,
                                      __bf16* __restrict__ Wt,
                                      int K, int Ncols) {
  int i = blockIdx.x * blockDim.x + threadIdx.x;
  if (i >= K * Ncols) return;
  int k = i / Ncols, n = i % Ncols;
  Wt[(size_t)n * K + k] = (__bf16)W[i];
}

// ---------------------------------------------------------------------------
// Generic bf16-WMMA GEMM: out[M x Ncols] = epi( (A+abias) @ W + bias )
// 128 threads = 4 waves; workgroup owns 16 rows; wave w owns col tiles w, w+4,...
// EPI: 0 = store, 1 = gelu(exact), 2 = + addsrc (residual)
// ---------------------------------------------------------------------------
template <int EPI>
__global__ void gemm_bf16_kernel(const float* __restrict__ A,
                                 const float* __restrict__ abias,
                                 const __bf16* __restrict__ Wt,
                                 const float* __restrict__ bias,
                                 const float* __restrict__ addsrc,
                                 float* __restrict__ out,
                                 int K, int Ncols) {
  int lane = threadIdx.x & 31;
  int wave = threadIdx.x >> 5;
  int rowBase = blockIdx.x * 16;
  int row = rowBase + (lane & 15);
  const float* Arow = A + (size_t)row * K;
  int nKs = K >> 5;
  int nCt = Ncols >> 4;
  int rshift = (lane & 16) ? 8 : 0;

  for (int ct = wave; ct < nCt; ct += 4) {
    v8f acc = {};
    const __bf16* Wcol = Wt + (size_t)(ct * 16 + (lane & 15)) * K;
    for (int ks = 0; ks < nKs; ++ks) {
      v16bf a = load_a_frag_f32(Arow, abias, ks, lane);
      v16bf b = load_frag_bf16(Wcol, ks, lane);
      acc = __builtin_amdgcn_wmma_f32_16x16x32_bf16(false, a, false, b,
                                                    (short)0, acc, false, false);
    }
    int col = ct * 16 + (lane & 15);
    float bv = bias[col];
#pragma unroll
    for (int r = 0; r < 8; ++r) {
      int orow = rowBase + r + rshift;
      float v = acc[r] + bv;
      if (EPI == 1) {
        v = 0.5f * v * (1.0f + erff(v * 0.70710678118f));
      } else if (EPI == 2) {
        v += addsrc[(size_t)orow * Ncols + col];
      }
      out[(size_t)orow * Ncols + col] = v;
    }
  }
}

// ---------------------------------------------------------------------------
// AdaGN part 1: group norm (8 groups of 16 ch) -> normed = GN(h)*g + b
// ---------------------------------------------------------------------------
__global__ void adagn_norm_kernel(const float* __restrict__ h_source,
                                  const float* __restrict__ gn_gamma,
                                  const float* __restrict__ gn_beta,
                                  float* __restrict__ normed, int N) {
  int t = threadIdx.x;
  int n = blockIdx.x * 2 + (t >> 7);
  int c = t & 127;
  if (n >= N) return;
  float x = h_source[(size_t)n * HID + c];
  float s = x;
#pragma unroll
  for (int m = 8; m >= 1; m >>= 1) s += __shfl_xor(s, m, 16);
  float mu = s * (1.0f / 16.0f);
  float d = x - mu;
  float q = d * d;
#pragma unroll
  for (int m = 8; m >= 1; m >>= 1) q += __shfl_xor(q, m, 16);
  float r = rsqrtf(q * (1.0f / 16.0f) + 1e-5f);
  normed[(size_t)n * HID + c] = d * r * gn_gamma[c] + gn_beta[c];
}

// AdaGN part 2: hmod = normed * (1 + gamma) + beta, style = [gamma|beta]
__global__ void adagn_mod_kernel(const float* __restrict__ normed,
                                 const float* __restrict__ style,
                                 float* __restrict__ hmod, int N) {
  int i = blockIdx.x * blockDim.x + threadIdx.x;
  if (i >= N * HID) return;
  int n = i >> 7, c = i & 127;
  float g = style[(size_t)n * 256 + c];
  float b = style[(size_t)n * 256 + 128 + c];
  hmod[i] = normed[i] * (1.0f + g) + b;
}

// ---------------------------------------------------------------------------
// Edge kernel: 16 edges per workgroup (128 thr = 4 waves).
// Wave 0 DMAs WeT (64KB bf16, column-major) into LDS with the Tensor Data
// Mover while all waves build edge_attr[16][256] (bf16) in LDS; then WMMA
// edge_attr @ We, epilogue z = leaky(xl[src]+xr[dst]+e), logit = <z, att>,
// global float atomic-max into m[dst][h].
// ---------------------------------------------------------------------------
__global__ void edge_logit_kernel(const int* __restrict__ ei,
                                  const float* __restrict__ pos,
                                  const float* __restrict__ hmod,
                                  const float* __restrict__ xl,
                                  const float* __restrict__ xr,
                                  const __bf16* __restrict__ WeT, // [128][256]
                                  const float* __restrict__ att,  // [4][32]
                                  float* __restrict__ logit,      // [E][4]
                                  float* __restrict__ mmax,       // [N][4]
                                  int E) {
  __shared__ __bf16 we_lds[128 * 256];   // 64KB staged weight tile
  __shared__ __bf16 attr[16][256];       // 8KB bf16 edge features
  __shared__ int    s_src[16], s_dst[16];
  __shared__ float  s_logit[16][4];
  int t = threadIdx.x;
  int lane = t & 31, wave = t >> 5;
  int e0 = blockIdx.x * 16;

#if HAVE_TDM
  if (wave == 0) {
    // TDM descriptor (CDNA5 ISA): 64KB as 8192 x 8B elements.
    unsigned long long ga = (unsigned long long)(uintptr_t)WeT;
    unsigned int lds_off = (unsigned int)(uintptr_t)&we_lds[0]; // flat[31:0] = LDS offset
    u32x4 g0;
    g0[0] = 1u;                                   // count=1, user descriptor
    g0[1] = lds_off;                              // lds_addr
    g0[2] = (unsigned int)ga;                     // global_addr[31:0]
    g0[3] = ((unsigned int)(ga >> 32) & 0x01FFFFFFu) | 0x80000000u; // addr[56:32] | type=2
    i32x8 g1;
    g1[0] = (int)(3u << 16);                      // data_size = 3 (8B), mask=0
    g1[1] = (int)((8192u & 0xFFFFu) << 16);       // tensor_dim0[15:0]
    g1[2] = (int)((8192u >> 16) | (1u << 16));    // tensor_dim0[31:16] | tensor_dim1[15:0]
    g1[3] = (int)(8192u << 16);                   // tensor_dim1[31:16]=0 | tile_dim0
    g1[4] = (int)1u;                              // tile_dim1 = 1, tile_dim2 = 0
    g1[5] = (int)8192u;                           // tensor_dim0_stride[31:0]
    g1[6] = 0;                                    // stride0[47:32] | stride1[15:0]
    g1[7] = 0;                                    // stride1[47:16]
    i32x4 g2 = {0, 0, 0, 0};
    i32x4 g3 = {0, 0, 0, 0};
#if defined(__clang_major__) && (__clang_major__ >= 23)
    i32x8 g4 = {0, 0, 0, 0, 0, 0, 0, 0};
    __builtin_amdgcn_tensor_load_to_lds(g0, g1, g2, g3, g4, 0);
#else
    __builtin_amdgcn_tensor_load_to_lds(g0, g1, g2, g3, 0);
#endif
  }
#else
  { // cooperative fallback copy (16K dwords over 128 threads)
    const unsigned int* s = (const unsigned int*)WeT;
    unsigned int* d = (unsigned int*)&we_lds[0];
    for (int i = t; i < 128 * 256 / 2; i += 128) d[i] = s[i];
  }
#endif

  if (t < 16) { s_src[t] = ei[e0 + t]; s_dst[t] = ei[E + e0 + t]; }
  if (t < 64) s_logit[t >> 2][t & 3] = 0.0f;
  __syncthreads();

  { // edge_attr: thread t -> edge el = t/8, channel chunk (t&7)*16
    int el = t >> 3, sub = t & 7;
    int src = s_src[el], dst = s_dst[el];
    float dx = pos[src * 2 + 0] - pos[dst * 2 + 0];
    float dy = pos[src * 2 + 1] - pos[dst * 2 + 1];
    float inv = 1.0f / fmaxf(dx * dx + dy * dy, 1e-8f);
    const float* hs = hmod + (size_t)src * HID;
    const float* hd = hmod + (size_t)dst * HID;
#pragma unroll
    for (int i = 0; i < 16; ++i) {
      int c = sub * 16 + i;
      float g = (hs[c] - hd[c]) * inv;
      attr[el][c]       = (__bf16)(-dy * g);
      attr[el][128 + c] = (__bf16)( dx * g);
    }
  }
#if HAVE_TDM
  if (wave == 0) __builtin_amdgcn_s_wait_tensorcnt(0);
#endif
  __syncthreads();

  int rshift = (lane & 16) ? 8 : 0;
  const __bf16* Arow = &attr[lane & 15][0];
  for (int ct = wave; ct < 8; ct += 4) {
    v8f acc = {};
    const __bf16* Wcol = &we_lds[(ct * 16 + (lane & 15)) * 256];
    for (int ks = 0; ks < 8; ++ks) { // K = 256
      v16bf a = load_frag_bf16(Arow, ks, lane);
      v16bf b = load_frag_bf16(Wcol, ks, lane);
      acc = __builtin_amdgcn_wmma_f32_16x16x32_bf16(false, a, false, b,
                                                    (short)0, acc, false, false);
    }
    int col = ct * 16 + (lane & 15);
    int head = ct >> 1;                 // 16-ch tile sits inside one 32-ch head
    float av = att[head * CHD + (col & 31)];
#pragma unroll
    for (int r = 0; r < 8; ++r) {
      int el = r + rshift;
      float v = acc[r] + xl[(size_t)s_src[el] * HID + col]
                       + xr[(size_t)s_dst[el] * HID + col];
      v = (v > 0.0f) ? v : 0.2f * v;    // leaky_relu
      v *= av;
#pragma unroll
      for (int m = 8; m >= 1; m >>= 1) v += __shfl_xor(v, m, 16);
      if ((lane & 15) == 0) atomicAdd(&s_logit[el][head], v);
    }
  }
  __syncthreads();

  if (t < 64) {
    int el = t >> 2, h = t & 3;
    float lg = s_logit[el][h];
    int e = e0 + el;
    logit[(size_t)e * 4 + h] = lg;
    float* addr = &mmax[(size_t)s_dst[el] * 4 + h];
    if (lg >= 0.0f) atomicMax((int*)addr, __float_as_int(lg));
    else            atomicMin((unsigned int*)addr, __float_as_uint(lg));
  }
}

// Softmax numerator + denominator scatter
__global__ void edge_softmax_num_kernel(const int* __restrict__ ei,
                                        const float* __restrict__ logit,
                                        const float* __restrict__ mmax,
                                        float* __restrict__ ex,
                                        float* __restrict__ denom, int E) {
  int i = blockIdx.x * blockDim.x + threadIdx.x;
  if (i >= E * 4) return;
  int e = i >> 2, h = i & 3;
  int dst = ei[E + e];
  float m = mmax[(size_t)dst * 4 + h];
  if (!__builtin_isfinite(m)) m = 0.0f;
  float v = expf(logit[i] - m);
  ex[i] = v;
  atomicAdd(&denom[(size_t)dst * 4 + h], v);
}

// msg = xl[src] * alpha; aggr[dst] += msg  (2 edges per 256-thread block)
__global__ void edge_aggregate_kernel(const int* __restrict__ ei,
                                      const float* __restrict__ ex,
                                      const float* __restrict__ denom,
                                      const float* __restrict__ xl,
                                      float* __restrict__ aggr, int E) {
  int t = threadIdx.x;
  int e = blockIdx.x * 2 + (t >> 7);
  if (e >= E) return;
  int c = t & 127;
  int src = ei[e], dst = ei[E + e];
  int h = c >> 5;
  float alpha = ex[(size_t)e * 4 + h] / (denom[(size_t)dst * 4 + h] + 1e-16f);
  atomicAdd(&aggr[(size_t)dst * HID + c], xl[(size_t)src * HID + c] * alpha);
}

// LayerNorm over 128 channels: 1 wave / node, 4 ch per lane
__global__ void layernorm_kernel(const float* __restrict__ x,
                                 const float* __restrict__ g,
                                 const float* __restrict__ b,
                                 float* __restrict__ out, int N) {
  int lane = threadIdx.x & 31;
  int n = blockIdx.x * 8 + (threadIdx.x >> 5);
  if (n >= N) return;
  const float* row = x + (size_t)n * HID;
  float v0 = row[lane], v1 = row[lane + 32], v2 = row[lane + 64], v3 = row[lane + 96];
  float s = v0 + v1 + v2 + v3;
#pragma unroll
  for (int m = 16; m >= 1; m >>= 1) s += __shfl_xor(s, m, 32);
  float mu = s * (1.0f / 128.0f);
  float d0 = v0 - mu, d1 = v1 - mu, d2 = v2 - mu, d3 = v3 - mu;
  float q = d0 * d0 + d1 * d1 + d2 * d2 + d3 * d3;
#pragma unroll
  for (int m = 16; m >= 1; m >>= 1) q += __shfl_xor(q, m, 32);
  float r = rsqrtf(q * (1.0f / 128.0f) + 1e-5f);
  float* orow = out + (size_t)n * HID;
  orow[lane]      = d0 * r * g[lane]      + b[lane];
  orow[lane + 32] = d1 * r * g[lane + 32] + b[lane + 32];
  orow[lane + 64] = d2 * r * g[lane + 64] + b[lane + 64];
  orow[lane + 96] = d3 * r * g[lane + 96] + b[lane + 96];
}

__global__ void init_m_denom_kernel(float* __restrict__ mmax,
                                    float* __restrict__ denom, int n4) {
  int i = blockIdx.x * blockDim.x + threadIdx.x;
  if (i >= n4) return;
  mmax[i] = -__builtin_inff();
  denom[i] = 0.0f;
}

__global__ void zero_kernel(float* __restrict__ p, int n) {
  int i = blockIdx.x * blockDim.x + threadIdx.x;
  if (i < n) p[i] = 0.0f;
}

// ---------------------------------------------------------------------------
// Host-side orchestration
// ---------------------------------------------------------------------------
extern "C" void kernel_launch(void* const* d_in, const int* in_sizes, int n_in,
                              void* d_out, int out_size, void* d_ws, size_t ws_size,
                              hipStream_t stream) {
  const float* h_target = (const float*)d_in[0];
  const float* h_source = (const float*)d_in[1];
  const float* pos      = (const float*)d_in[2];
  const float* t_emb    = (const float*)d_in[3];
  const int*   ei       = (const int*)  d_in[4];
  const float* gn_gamma = (const float*)d_in[5];
  const float* gn_beta  = (const float*)d_in[6];
  const float* fc_W     = (const float*)d_in[7];
  const float* fc_b     = (const float*)d_in[8];
  const float* Wl       = (const float*)d_in[9];
  const float* bl       = (const float*)d_in[10];
  const float* Wr       = (const float*)d_in[11];
  const float* br       = (const float*)d_in[12];
  const float* We       = (const float*)d_in[13];
  const float* att      = (const float*)d_in[14];
  const float* gat_bias = (const float*)d_in[15];
  const float* Wpost    = (const float*)d_in[16];
  const float* bpost    = (const float*)d_in[17];
  const float* ln_g     = (const float*)d_in[18];
  const float* ln_b     = (const float*)d_in[19];
  const float* W1       = (const float*)d_in[20];
  const float* b1       = (const float*)d_in[21];
  const float* W2       = (const float*)d_in[22];
  const float* b2       = (const float*)d_in[23];
  float* out = (float*)d_out;

  const int N = in_sizes[0] / HID;
  const int E = in_sizes[4] / 2;

  // --- workspace layout (lifetime-aliased) ---
  char* wsb = (char*)d_ws;
  __bf16* wt_fc   = (__bf16*)wsb;            // 256 x 128
  __bf16* wt_l    = wt_fc   + 256 * 128;     // 128 x 128
  __bf16* wt_r    = wt_l    + 128 * 128;
  __bf16* wt_e    = wt_r    + 128 * 128;     // 128 x 256
  __bf16* wt_post = wt_e    + 128 * 256;
  __bf16* wt_1    = wt_post + 128 * 128;
  __bf16* wt_2    = wt_1    + 128 * 128;

  float* fb   = (float*)(wsb + 512 * 1024);
  float* buf1 = fb;                               // N*128 : normed -> aggr -> ffn1
  float* buf2 = buf1 + (size_t)N * 256;           // N*256 : style -> xl|xr
  float* buf3 = buf2 + (size_t)N * 256;           // N*128 : hmod -> post
  float* buf4 = buf3 + (size_t)N * 128;           // max(N*128, E*8) : logit|ex -> ln
  size_t b4   = ((size_t)N * 128 > (size_t)E * 8) ? (size_t)N * 128 : (size_t)E * 8;
  float* mmax  = buf4 + b4;                       // N*4
  float* denom = mmax + (size_t)N * 4;            // N*4

  float* normed = buf1;  float* aggr = buf1;  float* ffn1 = buf1;
  float* style  = buf2;
  float* xl     = buf2;  float* xr   = buf2 + (size_t)N * HID;
  float* hmod   = buf3;  float* post = buf3;
  float* logit  = buf4;  float* exb  = buf4 + (size_t)E * 4;
  float* ln     = buf4;

  // 0) weights -> bf16 column-major
  {
    int thr = 256;
    transpose_bf16_kernel<<<(128 * 256 + thr - 1) / thr, thr, 0, stream>>>(fc_W, wt_fc, 128, 256);
    transpose_bf16_kernel<<<(128 * 128 + thr - 1) / thr, thr, 0, stream>>>(Wl, wt_l, 128, 128);
    transpose_bf16_kernel<<<(128 * 128 + thr - 1) / thr, thr, 0, stream>>>(Wr, wt_r, 128, 128);
    transpose_bf16_kernel<<<(256 * 128 + thr - 1) / thr, thr, 0, stream>>>(We, wt_e, 256, 128);
    transpose_bf16_kernel<<<(128 * 128 + thr - 1) / thr, thr, 0, stream>>>(Wpost, wt_post, 128, 128);
    transpose_bf16_kernel<<<(128 * 128 + thr - 1) / thr, thr, 0, stream>>>(W1, wt_1, 128, 128);
    transpose_bf16_kernel<<<(128 * 128 + thr - 1) / thr, thr, 0, stream>>>(W2, wt_2, 128, 128);
  }
  // softmax state init
  init_m_denom_kernel<<<(N * 4 + 255) / 256, 256, 0, stream>>>(mmax, denom, N * 4);

  // 1) GroupNorm
  adagn_norm_kernel<<<N / 2, 256, 0, stream>>>(h_source, gn_gamma, gn_beta, normed, N);
  // 2) style = t_emb @ fc_W + fc_b   (WMMA, Ncols=256)
  gemm_bf16_kernel<0><<<N / 16, 128, 0, stream>>>(t_emb, nullptr, wt_fc, fc_b, nullptr, style, 128, 256);
  // 3) hmod = normed*(1+gamma)+beta
  adagn_mod_kernel<<<(N * HID + 255) / 256, 256, 0, stream>>>(normed, style, hmod, N);
  // aggr buffer (aliases dead `normed`) -> zero
  zero_kernel<<<(N * HID + 255) / 256, 256, 0, stream>>>(aggr, N * HID);
  // 4) xl, xr   (WMMA)
  gemm_bf16_kernel<0><<<N / 16, 128, 0, stream>>>(hmod, nullptr, wt_l, bl, nullptr, xl, 128, 128);
  gemm_bf16_kernel<0><<<N / 16, 128, 0, stream>>>(hmod, nullptr, wt_r, br, nullptr, xr, 128, 128);
  // 5) per-edge WMMA transform (TDM-staged We) + logits + segment max
  edge_logit_kernel<<<E / 16, 128, 0, stream>>>(ei, pos, hmod, xl, xr, wt_e, att, logit, mmax, E);
  // 6) exp + denom
  edge_softmax_num_kernel<<<(E * 4 + 255) / 256, 256, 0, stream>>>(ei, logit, mmax, exb, denom, E);
  // 7) weighted scatter-add
  edge_aggregate_kernel<<<E / 2, 256, 0, stream>>>(ei, exb, denom, xl, aggr, E);
  // 8) post = (aggr + gat_bias) @ Wpost + bpost   (WMMA, A-side bias)
  gemm_bf16_kernel<0><<<N / 16, 128, 0, stream>>>(aggr, gat_bias, wt_post, bpost, nullptr, post, 128, 128);
  // 9) LayerNorm
  layernorm_kernel<<<N / 8, 256, 0, stream>>>(post, ln_g, ln_b, ln, N);
  // 10) ffn1 = gelu(ln @ W1 + b1)   (WMMA + exact gelu epilogue)
  gemm_bf16_kernel<1><<<N / 16, 128, 0, stream>>>(ln, nullptr, wt_1, b1, nullptr, ffn1, 128, 128);
  // 11) out = h_target + ffn1 @ W2 + b2   (WMMA + residual epilogue)
  gemm_bf16_kernel<2><<<N / 16, 128, 0, stream>>>(ffn1, nullptr, wt_2, b2, h_target, out, 128, 128);
}
